// SputnikAGNNLayer_60241211293936
// MI455X (gfx1250) — matrix operand order repaired and checked by reading, not compiled, not run on hardware
//
#include <hip/hip_runtime.h>
#include <hip/hip_bf16.h>

#define FDIM 128
#define DEG  16
#define WPB  8   // waves (nodes) per 256-thread block

typedef __attribute__((ext_vector_type(16))) _Float16 v16h;
typedef __attribute__((ext_vector_type(8)))  _Float16 v8h;
typedef __attribute__((ext_vector_type(4)))  _Float16 v4h;
typedef __attribute__((ext_vector_type(8)))  float    v8f;
typedef __attribute__((ext_vector_type(4)))  float    v4f;

// DPP lane exchange (VALU pipe, no LDS round-trip). All lanes active.
#define DPPF(v, ctrl) \
  __uint_as_float(__builtin_amdgcn_mov_dpp(__float_as_uint(v), (ctrl), 0xf, 0xf, true))
#define DPP_QUAD_XOR1 0xB1   // quad_perm(1,0,3,2)
#define DPP_QUAD_XOR2 0x4E   // quad_perm(2,3,0,1)
#define DPP_HALF_MIRR 0x141  // row_half_mirror (xor 7 within 8)
#define DPP_ROW_MIRR  0x140  // row_mirror (xor 15 within 16)

// ds_swizzle for the one cross-row (xor 16) step in the norms reduction
#define SWZ_XOR16(v) \
  __uint_as_float(__builtin_amdgcn_ds_swizzle(__float_as_uint(v), (0x1f | (16 << 10))))

static __device__ __forceinline__ float bcast_lane(float v, int lane) {
  return __uint_as_float(__builtin_amdgcn_readlane(__float_as_uint(v), lane));
}

// butterfly reduce over each 16-lane row; every lane ends with the row result
static __device__ __forceinline__ float red16_max(float x) {
  x = fmaxf(x, DPPF(x, DPP_QUAD_XOR1));
  x = fmaxf(x, DPPF(x, DPP_QUAD_XOR2));
  x = fmaxf(x, DPPF(x, DPP_HALF_MIRR));   // quads uniform -> xor7 == xor4
  x = fmaxf(x, DPPF(x, DPP_ROW_MIRR));    // 8-groups uniform -> xor15 == xor8
  return x;
}
static __device__ __forceinline__ float red16_sum(float x) {
  x += DPPF(x, DPP_QUAD_XOR1);
  x += DPPF(x, DPP_QUAD_XOR2);
  x += DPPF(x, DPP_HALF_MIRR);
  x += DPPF(x, DPP_ROW_MIRR);
  return x;
}

// ---------------- pass 1: per-node inverse norms -> d_ws ----------------
__global__ __launch_bounds__(256) void agnn_norms(const float* __restrict__ x,
                                                  float* __restrict__ rn, int n) {
  const int wave = threadIdx.x >> 5;
  const int lane = threadIdx.x & 31;
  const int node = blockIdx.x * WPB + wave;
  if (node >= n) return;                       // wave-uniform
  v4f v = ((const v4f*)(x + (size_t)node * FDIM))[lane];
  float ssq = v[0]*v[0] + v[1]*v[1] + v[2]*v[2] + v[3]*v[3];
  ssq = red16_sum(ssq);                        // in-row (VALU DPP)
  ssq += SWZ_XOR16(ssq);                       // cross-row
  if (lane == 0) rn[node] = rsqrtf(ssq + 1e-12f);
}

// ---------------- pass 2: WMMA scores + softmax + aggregation ----------------
__global__ __launch_bounds__(256) void agnn_main(const float* __restrict__ x,
                                                 const int*   __restrict__ col_id,
                                                 const float* __restrict__ beta,
                                                 const float* __restrict__ rn,
                                                 float* __restrict__ out, int n) {
  // per-wave private LDS staging (raw features converted to f16), +8-half pad
  __shared__ _Float16 sNbr[WPB][DEG][FDIM + 8];
  __shared__ _Float16 sQ[WPB][FDIM + 8];

  const int wave = threadIdx.x >> 5;
  const int lane = threadIdx.x & 31;
  const int node = blockIdx.x * WPB + wave;
  if (node >= n) return;                        // wave-uniform; EXEC stays all-ones
  const int c    = lane & 15;                   // WMMA column / neighbor id
  const int half = lane >> 4;                   // lane-group selector for K slots

  // --- per-lane neighbor metadata: lane handles neighbor c
  const int   colv = col_id[node * DEG + c];    // one vector load (lanes 16..31 mirror)
  const float rnjv = rn[colv];                  // one gather: inverse norm of neighbor c
  const float rni  = rn[node];
  const float bta  = beta[0];

  // --- own row -> sQ (f16)
  v4f q = ((const v4f*)(x + (size_t)node * FDIM))[lane];
  ((v4h*)&sQ[wave][0])[lane] = __builtin_convertvector(q, v4h);

  // --- gather 16 raw neighbor rows: registers (fp32, for aggregation) + LDS (f16, WMMA B)
  v4f r[DEG];
  #pragma unroll
  for (int j = 0; j < DEG; ++j) {
    const int col = __builtin_amdgcn_readlane(colv, j);   // SGPR base, no mem traffic
    r[j] = ((const v4f*)(x + (size_t)col * FDIM))[lane];
    ((v4h*)&sNbr[wave][j][0])[lane] = __builtin_convertvector(r[j], v4h);
  }

  // --- A fragments: q replicated in all 16 rows.
  // 16-bit A 16x32 layout: lanes<16 hold K={t*32+0..7, t*32+16..23}, lanes>=16 hold +8.
  const v8h* qp = (const v8h*)&sQ[wave][0];
  v16h afrag[4];
  #pragma unroll
  for (int t = 0; t < 4; ++t) {
    v8h lo = qp[t * 4 + half];
    v8h hi = qp[t * 4 + half + 2];
    afrag[t] = __builtin_shufflevector(lo, hi, 0,1,2,3,4,5,6,7,8,9,10,11,12,13,14,15);
  }

  // --- K-chained WMMA: D = A(16x128) * B(128x16), column c = neighbor c
  v8f acc = {};
  const v8h* np = (const v8h*)&sNbr[wave][c][0];
  #pragma unroll
  for (int t = 0; t < 4; ++t) {
    // B 32x16 layout (matches A): lanes<16 hold K={t*32+0..7, t*32+16..23}, lanes>=16 hold +8.
    v8h b0 = np[t * 4 + half];
    v8h b1 = np[t * 4 + half + 2];
    v16h bfrag = __builtin_shufflevector(b0, b1, 0,1,2,3,4,5,6,7,8,9,10,11,12,13,14,15);
    acc = __builtin_amdgcn_wmma_f32_16x16x32_f16(
        /*neg_a=*/false, afrag[t], /*neg_b=*/false, bfrag,
        /*c_mod=*/(short)0, acc, /*reuse_a=*/false, /*reuse_b=*/false);
  }

  // acc[0] = raw dot(x_i, x_nbr[c]) in every lane (all D rows identical)
  const float e = bta * rni * rnjv * acc[0];

  // --- softmax over the 16 neighbors, entirely in the VALU pipe (DPP butterflies)
  const float m     = red16_max(e);
  const float ee    = __expf(e - m);
  const float ssum  = red16_sum(ee);
  const float alpha = ee / ssum;

  // --- weighted aggregation of raw fp32 neighbor rows (register-cached)
  v4f o = {0.0f, 0.0f, 0.0f, 0.0f};
  #pragma unroll
  for (int j = 0; j < DEG; ++j) {
    const float a = bcast_lane(alpha, j);       // v_readlane -> SGPR fma operand
    o += a * r[j];
  }
  // streaming store: out is never re-read; keep L2 for x
  __builtin_nontemporal_store(o, &((v4f*)(out + (size_t)node * FDIM))[lane]);
}

// ---------------- host launcher ----------------
extern "C" void kernel_launch(void* const* d_in, const int* in_sizes, int n_in,
                              void* d_out, int out_size, void* d_ws, size_t ws_size,
                              hipStream_t stream) {
  const float* x      = (const float*)d_in[0];   // [N,128] f32
  const int*   col_id = (const int*)  d_in[3];   // [E] i32
  const float* beta   = (const float*)d_in[4];   // [1] f32
  float*       out    = (float*)d_out;           // [N,128] f32
  float*       rn     = (float*)d_ws;            // [N] f32 inverse norms

  const int n      = in_sizes[0] / FDIM;         // 50000
  const int blocks = (n + WPB - 1) / WPB;        // 6250

  agnn_norms<<<blocks, 256, 0, stream>>>(x, rn, n);
  agnn_main <<<blocks, 256, 0, stream>>>(x, col_id, beta, rn, out, n);
}
